// ContrastiveMambaEncoder_44633300140426
// MI455X (gfx1250) — matrix operand order repaired and checked by reading, not compiled, not run on hardware
//
#include <hip/hip_runtime.h>
#include <math.h>

// ---------------- problem constants (from reference) ----------------
#define B_    8
#define L_    1024
#define DM_   1024
#define H_    32
#define P_    64
#define NS_   128            // d_state
#define KC_   4              // d_conv
#define DI_   2048           // H*P
#define DXBC_ 2304           // DI + 2*NS
#define DINP_ 4384           // 2*DI + 2*NS + H
#define ED_   2048
#define MROWS_ 8192          // B*L

// ---------------- WMMA types ----------------
typedef __attribute__((ext_vector_type(16))) __bf16 v16bf;
typedef __attribute__((ext_vector_type(8)))  float  v8f;

union Frag { v16bf v; uint4 q[2]; };

__device__ __forceinline__ unsigned short f2bf(float f) {
  unsigned int u = __float_as_uint(f);
  u = u + 0x7FFFu + ((u >> 16) & 1u);   // round-to-nearest-even
  return (unsigned short)(u >> 16);
}

// ---------------- gfx1250 async global->LDS helpers (inline asm) ----------------
__device__ __forceinline__ unsigned lds_off_of(const void* p) {
  return (unsigned)(unsigned long long)(const __attribute__((address_space(3))) char*)p;
}
__device__ __forceinline__ void async_copy_b128(unsigned lds_off, const void* gptr) {
  asm volatile("global_load_async_to_lds_b128 %0, %1, off"
               :: "v"(lds_off), "v"(gptr) : "memory");
}
__device__ __forceinline__ void async_copy_b32(unsigned lds_off, const void* gptr) {
  asm volatile("global_load_async_to_lds_b32 %0, %1, off"
               :: "v"(lds_off), "v"(gptr) : "memory");
}
__device__ __forceinline__ void wait_async0() {
  asm volatile("s_wait_asynccnt 0" ::: "memory");
}

// ---------------- f32 -> bf16 elementwise convert (vectorized) ----------------
__global__ __launch_bounds__(256)
void cvt_bf16(const float* __restrict__ in, unsigned short* __restrict__ out, long n) {
  long i = ((long)blockIdx.x * 256 + threadIdx.x) * 4;
  if (i >= n) return;
  float4 v = *reinterpret_cast<const float4*>(&in[i]);
  unsigned lo = (unsigned)f2bf(v.x) | ((unsigned)f2bf(v.y) << 16);
  unsigned hi = (unsigned)f2bf(v.z) | ((unsigned)f2bf(v.w) << 16);
  *reinterpret_cast<uint2*>(&out[i]) = make_uint2(lo, hi);
}

// ---------------- W[K][N] f32 -> Wt[N][K] bf16 (tiled transpose) ----------------
__global__ __launch_bounds__(256)
void transp_cvt(const float* __restrict__ W, unsigned short* __restrict__ Wt,
                int Kdim, int Ndim) {
  __shared__ float t[32][33];
  const int k0 = blockIdx.y * 32, n0 = blockIdx.x * 32;
  const int tx = threadIdx.x & 31, ty = threadIdx.x >> 5;   // ty: 0..7
#pragma unroll
  for (int i = 0; i < 4; ++i) {
    int k = ty + i * 8;
    int n = n0 + tx;
    t[k][tx] = (n < Ndim) ? W[(long)(k0 + k) * Ndim + n] : 0.f;
  }
  __syncthreads();
#pragma unroll
  for (int i = 0; i < 4; ++i) {
    int nrow = ty + i * 8;
    if (n0 + nrow < Ndim)
      Wt[(long)(n0 + nrow) * Kdim + k0 + tx] = f2bf(t[tx][nrow]);
  }
}

// ---------------- bf16 WMMA GEMM: C = A[MxK] * Wt[NxK]^T (+bias) ----------------
#define TM 128
#define TN 64
#define TK 64
#define LDB 72   // padded LDS row stride (ushorts): 144B, 16B-aligned, conflict-free

__global__ __launch_bounds__(256)
void gemm_bias_wmma(const unsigned short* __restrict__ A,
                    const unsigned short* __restrict__ Wt,
                    const float* __restrict__ bias, float* __restrict__ C,
                    int Kdim, int Ndim) {
  __shared__ __align__(16) unsigned short As[2][TM * LDB];
  __shared__ __align__(16) unsigned short Bs[2][TN * LDB];

  const int tid  = threadIdx.x;
  const int lane = tid & 31;
  const int wave = tid >> 5;
  const int wm   = (wave & 3) * 32;   // wave M offset in macro-tile
  const int wn   = (wave >> 2) * 32;  // wave N offset in macro-tile
  const int bm   = blockIdx.y * TM;
  const int bn   = blockIdx.x * TN;

  v8f acc[2][2];
  acc[0][0] = {}; acc[0][1] = {}; acc[1][0] = {}; acc[1][1] = {};

  // async tile staging: raw bf16 byte copies, 16B per lane per op
  auto stageA = [&](int k0, int buf) {
#pragma unroll
    for (int i = 0; i < 4; ++i) {
      int li = tid + i * 256;            // 1024 chunks of 8 ushorts
      int row = li >> 3, ch = (li & 7) * 8;
      async_copy_b128(lds_off_of(&As[buf][row * LDB + ch]),
                      &A[(long)(bm + row) * Kdim + k0 + ch]);
    }
  };
  auto stageB = [&](int k0, int buf) {
#pragma unroll
    for (int i = 0; i < 2; ++i) {
      int li = tid + i * 256;            // 512 chunks of 8 ushorts
      int row = li >> 3, ch = (li & 7) * 8;
      int gr = bn + row;
      if (gr >= Ndim) gr = Ndim - 1;     // clamp; garbage cols masked at store
      async_copy_b128(lds_off_of(&Bs[buf][row * LDB + ch]),
                      &Wt[(long)gr * Kdim + k0 + ch]);
    }
  };

  const int kTiles = Kdim / TK;
  stageA(0, 0);
  stageB(0, 0);
  wait_async0();
  __syncthreads();

  const int lo  = lane & 15;
  const int kb  = (lane >> 4) * 8;    // A: k-base 0 or 8   (ISA 16-bit A 16x32)
  const int kof = (lane >> 4) * 8;    // B: k-base 0 or 8   (ISA 16-bit B 32x16)

  for (int kt = 0; kt < kTiles; ++kt) {
    const int buf = kt & 1;
    if (kt + 1 < kTiles) {
      stageA((kt + 1) * TK, buf ^ 1);
      stageB((kt + 1) * TK, buf ^ 1);
    }
    if (kt + 2 < kTiles)  // keep a far prefetch on the global path
      __builtin_prefetch(&A[(long)(bm + (tid >> 1)) * Kdim + (kt + 2) * TK], 0, 1);

    const unsigned short* as = As[buf];
    const unsigned short* bs = Bs[buf];
#pragma unroll
    for (int s = 0; s < 2; ++s) {       // two 16x16x32 k-steps per 64-wide tile
      Frag afr[2], bfr[2];
#pragma unroll
      for (int mt = 0; mt < 2; ++mt) {
        const unsigned short* p = &as[(wm + mt * 16 + lo) * LDB + s * 32 + kb];
        afr[mt].q[0] = *reinterpret_cast<const uint4*>(p);
        afr[mt].q[1] = *reinterpret_cast<const uint4*>(p + 16);
      }
#pragma unroll
      for (int nt = 0; nt < 2; ++nt) {
        const unsigned short* p = &bs[(wn + nt * 16 + lo) * LDB + s * 32 + kof];
        bfr[nt].q[0] = *reinterpret_cast<const uint4*>(p);
        bfr[nt].q[1] = *reinterpret_cast<const uint4*>(p + 16);
      }
#pragma unroll
      for (int mt = 0; mt < 2; ++mt)
#pragma unroll
        for (int nt = 0; nt < 2; ++nt)
          acc[mt][nt] = __builtin_amdgcn_wmma_f32_16x16x32_bf16(
              false, afr[mt].v, false, bfr[nt].v, (short)0, acc[mt][nt],
              false, false);
    }
    wait_async0();     // our async writes into buf^1 landed
    __syncthreads();   // everyone done reading buf & done staging buf^1
  }

  // epilogue: C/D layout — VGPR r: lanes0-15 M=r, lanes16-31 M=8+r; N=lane&15
  const int n_lo = lane & 15;
  const int m_hi = (lane >> 4) * 8;
#pragma unroll
  for (int mt = 0; mt < 2; ++mt) {
#pragma unroll
    for (int nt = 0; nt < 2; ++nt) {
      int col = bn + wn + nt * 16 + n_lo;
      if (col >= Ndim) continue;
      float bv = bias ? bias[col] : 0.f;
#pragma unroll
      for (int r = 0; r < 8; ++r) {
        int row = bm + wm + mt * 16 + m_hi + r;
        C[(long)row * Ndim + col] = acc[mt][nt][r] + bv;
      }
    }
  }
}

// ---------------- causal depthwise conv1d (K=4) + SiLU ----------------
__global__ __launch_bounds__(256)
void conv_silu(const float* __restrict__ zx, const float* __restrict__ conv_w,
               const float* __restrict__ conv_b, float* __restrict__ convout) {
  long idx = (long)blockIdx.x * 256 + threadIdx.x;   // B*L*DXBC
  int c = (int)(idx % DXBC_);
  long bl = idx / DXBC_;
  int l = (int)(bl % L_);
  float acc = conv_b[c];
#pragma unroll
  for (int k = 0; k < KC_; ++k) {
    int ls = l - (KC_ - 1) + k;
    if (ls >= 0)
      acc = fmaf(zx[(bl - l + ls) * DINP_ + DI_ + c], conv_w[c * KC_ + k], acc);
  }
  acc = acc / (1.f + __expf(-acc));   // SiLU
  convout[idx] = acc;
}

// ---------------- SSM sequential scan, one block per (b,h) ----------------
// B/C rows double-buffered via async global->LDS; the 4 state-chunk partials
// for each p live in adjacent lanes -> reduce with lane shuffles, one barrier
// per timestep (the double-buffer publish).
__global__ __launch_bounds__(256)
void ssm_scan(const float* __restrict__ zx, const float* __restrict__ conv,
              const float* __restrict__ dt_bias, const float* __restrict__ A_log,
              const float* __restrict__ Dv, float* __restrict__ yraw) {
  const int b = blockIdx.x >> 5;     // H_=32
  const int h = blockIdx.x & 31;
  const int tid = threadIdx.x;
  const int p  = tid >> 2;           // 0..63 (8 p per wave)
  const int nc = tid & 3;            // n-chunk, adjacent lanes
  const int n0 = nc * 32;
  __shared__ __align__(16) float BC[2][2 * NS_];
  __shared__ float sc[2][2];
  float hreg[32];
#pragma unroll
  for (int j = 0; j < 32; ++j) hreg[j] = 0.f;
  const float negA = -__expf(A_log[h]);
  const float Dh   = Dv[h];
  const float dtb  = dt_bias[h];

  // stage t = 0
  async_copy_b32(lds_off_of(&BC[0][tid]),
                 &conv[((long)b * L_) * DXBC_ + DI_ + tid]);
  if (tid == 0) {
    float xr  = zx[((long)b * L_) * DINP_ + (DI_ + DXBC_) + h] + dtb;
    float dts = (xr > 20.f) ? xr : log1pf(__expf(xr));
    sc[0][0] = dts;
    sc[0][1] = __expf(negA * dts);
  }
  wait_async0();
  __syncthreads();

  for (int t = 0; t < L_; ++t) {
    const long base = (long)b * L_ + t;
    const int buf = t & 1;
    if (t + 1 < L_) {   // prefetch next step's B/C rows + scalars
      async_copy_b32(lds_off_of(&BC[buf ^ 1][tid]),
                     &conv[(base + 1) * DXBC_ + DI_ + tid]);
      if (tid == 0) {
        float xr  = zx[(base + 1) * DINP_ + (DI_ + DXBC_) + h] + dtb;
        float dts = (xr > 20.f) ? xr : log1pf(__expf(xr));
        sc[buf ^ 1][0] = dts;
        sc[buf ^ 1][1] = __expf(negA * dts);
      }
    }
    float xp = conv[base * DXBC_ + h * P_ + p];
    const float dts = sc[buf][0], dAv = sc[buf][1];
    const float cx = dts * xp;
    const float* Bl = &BC[buf][0];
    const float* Cl = &BC[buf][NS_];
    float part = 0.f;
#pragma unroll
    for (int j = 0; j < 32; ++j) {
      hreg[j] = fmaf(hreg[j], dAv, cx * Bl[n0 + j]);
      part    = fmaf(hreg[j], Cl[n0 + j], part);
    }
    // reduce the 4 n-chunk partials across adjacent lanes (no barrier)
    part += __shfl_xor(part, 1, 32);
    part += __shfl_xor(part, 2, 32);
    if (nc == 0)
      yraw[base * DI_ + h * P_ + p] = part + Dh * xp;
    wait_async0();     // next-step async writes landed
    __syncthreads();   // publish BC[buf^1], sc[buf^1]
  }
}

// ---------------- gate (silu(z)) + RMSNorm over DI ----------------
__global__ __launch_bounds__(256)
void gate_norm(const float* __restrict__ yraw, const float* __restrict__ zx,
               const float* __restrict__ normw, float* __restrict__ out) {
  const long row = blockIdx.x;
  const int tid = threadIdx.x;
  float v[8];
  float ss = 0.f;
#pragma unroll
  for (int i = 0; i < 8; ++i) {
    int d = tid + i * 256;
    float z = zx[row * DINP_ + d];
    float g = yraw[row * DI_ + d] * (z / (1.f + __expf(-z)));
    v[i] = g;
    ss = fmaf(g, g, ss);
  }
  __shared__ float red[256];
  red[tid] = ss;
  __syncthreads();
  for (int s = 128; s > 0; s >>= 1) {
    if (tid < s) red[tid] += red[tid + s];
    __syncthreads();
  }
  const float inv = rsqrtf(red[0] / (float)DI_ + 1e-5f);
#pragma unroll
  for (int i = 0; i < 8; ++i) {
    int d = tid + i * 256;
    out[row * DI_ + d] = v[i] * inv * normw[d];
  }
}

// ---------------- attention score: emb . w_attn + b ----------------
__global__ __launch_bounds__(256)
void attn_score(const float* __restrict__ emb, const float* __restrict__ w_attn,
                const float* __restrict__ b_attn, float* __restrict__ scores) {
  const long row = blockIdx.x;
  const int tid = threadIdx.x;
  float acc = 0.f;
#pragma unroll
  for (int i = 0; i < 4; ++i) {
    int d = tid + i * 256;
    acc = fmaf(emb[row * DM_ + d], w_attn[d], acc);
  }
  __shared__ float red[256];
  red[tid] = acc;
  __syncthreads();
  for (int s = 128; s > 0; s >>= 1) {
    if (tid < s) red[tid] += red[tid + s];
    __syncthreads();
  }
  if (tid == 0) scores[row] = red[0] + b_attn[0];
}

// ---------------- softmax over L (weights in global), one block per batch ----------------
__global__ __launch_bounds__(256)
void softmax_weights(const float* __restrict__ scores, float* __restrict__ wts) {
  const int b = blockIdx.x;
  const int tid = threadIdx.x;
  __shared__ float w[L_];
  __shared__ float red[256];
#pragma unroll
  for (int i = 0; i < 4; ++i) w[tid + i * 256] = scores[b * L_ + tid + i * 256];
  __syncthreads();
  float m = -3.4e38f;
#pragma unroll
  for (int i = 0; i < 4; ++i) m = fmaxf(m, w[tid + i * 256]);
  red[tid] = m;
  __syncthreads();
  for (int s = 128; s > 0; s >>= 1) {
    if (tid < s) red[tid] = fmaxf(red[tid], red[tid + s]);
    __syncthreads();
  }
  m = red[0];
  __syncthreads();
  float ssum = 0.f;
#pragma unroll
  for (int i = 0; i < 4; ++i) {
    float e = __expf(w[tid + i * 256] - m);
    w[tid + i * 256] = e;
    ssum += e;
  }
  red[tid] = ssum;
  __syncthreads();
  for (int s = 128; s > 0; s >>= 1) {
    if (tid < s) red[tid] += red[tid + s];
    __syncthreads();
  }
  const float inv = 1.f / red[0];
  __syncthreads();
#pragma unroll
  for (int i = 0; i < 4; ++i)
    wts[b * L_ + tid + i * 256] = w[tid + i * 256] * inv;
}

// ---------------- weighted pooling: 32 L-chunks per batch, f32 atomic adds ----------------
__global__ __launch_bounds__(256)
void pool_accum(const float* __restrict__ emb, const float* __restrict__ wts,
                float* __restrict__ pooled) {
  const int blk = blockIdx.x;         // B_*32 blocks
  const int b  = blk >> 5;
  const int l0 = (blk & 31) * 32;
  const int tid = threadIdx.x;
  float a0 = 0.f, a1 = 0.f, a2 = 0.f, a3 = 0.f;
  for (int i = 0; i < 32; ++i) {
    int l = l0 + i;
    float wt = wts[b * L_ + l];
    const float* e = &emb[((long)b * L_ + l) * DM_];
    a0 = fmaf(e[tid], wt, a0);
    a1 = fmaf(e[tid + 256], wt, a1);
    a2 = fmaf(e[tid + 512], wt, a2);
    a3 = fmaf(e[tid + 768], wt, a3);
  }
  unsafeAtomicAdd(&pooled[b * DM_ + tid], a0);        // global_atomic_add_f32
  unsafeAtomicAdd(&pooled[b * DM_ + tid + 256], a1);
  unsafeAtomicAdd(&pooled[b * DM_ + tid + 512], a2);
  unsafeAtomicAdd(&pooled[b * DM_ + tid + 768], a3);
}

// ---------------- final: pooled @ w_emb + b_emb -> [B, ED] ----------------
__global__ __launch_bounds__(256)
void final_proj(const float* __restrict__ pooled, const float* __restrict__ w_emb,
                const float* __restrict__ b_emb, float* __restrict__ out) {
  const int idx = blockIdx.x * 256 + threadIdx.x;  // B*ED = 16384
  const int b = idx >> 11;                         // ED=2048
  const int e = idx & (ED_ - 1);
  float acc = b_emb[e];
  for (int k = 0; k < DM_; ++k)
    acc = fmaf(pooled[b * DM_ + k], w_emb[(long)k * ED_ + e], acc);
  out[idx] = acc;
}

// ---------------- host launcher ----------------
extern "C" void kernel_launch(void* const* d_in, const int* in_sizes, int n_in,
                              void* d_out, int out_size, void* d_ws, size_t ws_size,
                              hipStream_t stream) {
  const float* bert   = (const float*)d_in[0];
  const float* w_pb   = (const float*)d_in[1];
  const float* b_pb   = (const float*)d_in[2];
  const float* w_in   = (const float*)d_in[3];
  const float* conv_w = (const float*)d_in[4];
  const float* conv_b = (const float*)d_in[5];
  const float* dt_b   = (const float*)d_in[6];
  const float* A_log  = (const float*)d_in[7];
  const float* Dv     = (const float*)d_in[8];
  const float* norm_w = (const float*)d_in[9];
  const float* w_outp = (const float*)d_in[10];
  const float* w_attn = (const float*)d_in[11];
  const float* b_attn = (const float*)d_in[12];
  const float* w_emb  = (const float*)d_in[13];
  const float* b_emb  = (const float*)d_in[14];

  char* wp = (char*)d_ws;
  auto alloc = [&](size_t bytes) {
    char* r = wp;
    wp += (bytes + 255) & ~(size_t)255;
    return r;
  };
  float* x      = (float*)alloc((size_t)MROWS_ * DM_ * 4);    // also reused as emb
  float* zx     = (float*)alloc((size_t)MROWS_ * DINP_ * 4);
  float* convb  = (float*)alloc((size_t)MROWS_ * DXBC_ * 4);
  float* yraw   = (float*)alloc((size_t)MROWS_ * DI_ * 4);
  float* ynorm  = (float*)alloc((size_t)MROWS_ * DI_ * 4);
  float* scores = (float*)alloc((size_t)MROWS_ * 4);
  float* wts    = (float*)alloc((size_t)MROWS_ * 4);
  float* pooled = (float*)alloc((size_t)B_ * DM_ * 4);
  unsigned short* abf    = (unsigned short*)alloc((size_t)MROWS_ * DI_ * 2);  // max K
  unsigned short* wpbt   = (unsigned short*)alloc((size_t)DM_ * DM_ * 2);
  unsigned short* wint   = (unsigned short*)alloc((size_t)DINP_ * DM_ * 2);
  unsigned short* woutpt = (unsigned short*)alloc((size_t)DI_ * DM_ * 2);
  float* emb = x;

  dim3 blk(256);

  // weight transposes (bf16, [N][K])
  transp_cvt<<<dim3(DM_ / 32, 1024 / 32), blk, 0, stream>>>(w_pb, wpbt, 1024, DM_);
  transp_cvt<<<dim3(DINP_ / 32, 1024 / 32), blk, 0, stream>>>(w_in, wint, 1024, DINP_);
  transp_cvt<<<dim3(DM_ / 32, DI_ / 32), blk, 0, stream>>>(w_outp, woutpt, DI_, DM_);

  // 1) x = bert_out @ w_pb + b_pb      [8192,1024]x[1024,1024]
  cvt_bf16<<<(MROWS_ * DM_) / 1024, blk, 0, stream>>>(bert, abf, (long)MROWS_ * DM_);
  gemm_bias_wmma<<<dim3(DM_ / TN, MROWS_ / TM), blk, 0, stream>>>(
      abf, wpbt, b_pb, x, 1024, DM_);

  // 2) zxbcdt = x @ w_in               [8192,1024]x[1024,4384]
  cvt_bf16<<<(MROWS_ * DM_) / 1024, blk, 0, stream>>>(x, abf, (long)MROWS_ * DM_);
  gemm_bias_wmma<<<dim3((DINP_ + TN - 1) / TN, MROWS_ / TM), blk, 0, stream>>>(
      abf, wint, nullptr, zx, 1024, DINP_);

  // 3) depthwise causal conv + SiLU
  conv_silu<<<(MROWS_ * DXBC_) / 256, blk, 0, stream>>>(zx, conv_w, conv_b, convb);

  // 4) SSM scan (B*H blocks, async double-buffered staging, shuffle reduce)
  ssm_scan<<<B_ * H_, blk, 0, stream>>>(zx, convb, dt_b, A_log, Dv, yraw);

  // 5) gate + RMSNorm
  gate_norm<<<MROWS_, blk, 0, stream>>>(yraw, zx, norm_w, ynorm);

  // 6) emb = ynorm @ w_outp            [8192,2048]x[2048,1024]
  cvt_bf16<<<(MROWS_ * DI_) / 1024, blk, 0, stream>>>(ynorm, abf, (long)MROWS_ * DI_);
  gemm_bias_wmma<<<dim3(DM_ / TN, MROWS_ / TM), blk, 0, stream>>>(
      abf, woutpt, nullptr, emb, DI_, DM_);

  // 7) attention scores + softmax weights
  attn_score<<<MROWS_, blk, 0, stream>>>(emb, w_attn, b_attn, scores);
  softmax_weights<<<B_, blk, 0, stream>>>(scores, wts);

  // 8) pooling (zero-init + parallel atomic accumulation)
  hipMemsetAsync(pooled, 0, (size_t)B_ * DM_ * 4, stream);
  pool_accum<<<B_ * 32, blk, 0, stream>>>(emb, wts, pooled);

  // 9) final embedding projection
  final_proj<<<(B_ * ED_) / 256, blk, 0, stream>>>(pooled, w_emb, b_emb, (float*)d_out);

  (void)in_sizes; (void)n_in; (void)out_size; (void)ws_size;
}
